// GRFExactAttention_11398843203808
// MI455X (gfx1250) — compile-verified, matrix operand order checked by reading, not run on hardware
//
#include <hip/hip_runtime.h>
#include <hip/hip_bf16.h>

// ---------------------------------------------------------------------------
// GRF exact attention for MI455X (gfx1250), wave32 + WMMA f16->f32.
//
// All matrix operands staged in f16 with layouts chosen so every WMMA fragment
// load is two contiguous 16-byte loads (global_load_b128 / ds_load_b128):
//   0) cvt x -> f16 ; build w_qkv^T, w_out^T, mask^T in f16
//   1) qkv = x @ w_qkv            (WMMA)  -> Q^T/K^T/V^T [bh,d,n], relu+eps q,k
//   2) Qd = Q + 0.1*mask^T@Q      (WMMA)  -> row-major [bh,n,d]   (same for K)
//   3) fused masked linear attention: 8 waves/block share one (b,h); K/V tiles
//      staged into LDS with GLOBAL_LOAD_ASYNC_TO_LDS_B128 (ASYNCcnt)
//   4) out = O @ w_out + b_out    (WMMA)  -> f32 d_out
// ---------------------------------------------------------------------------

typedef __attribute__((ext_vector_type(16))) _Float16 v16h;
typedef __attribute__((ext_vector_type(8)))  _Float16 v8h;
typedef __attribute__((ext_vector_type(8)))  float    v8f;

// async-copy builtin pointer types: AS1/AS3 pointers to int4 vectors
typedef int v4i_ __attribute__((vector_size(16)));
typedef __attribute__((address_space(1))) v4i_ gv4i;
typedef __attribute__((address_space(3))) v4i_ sv4i;

#define WMMA_F16(A, B, C) \
  __builtin_amdgcn_wmma_f32_16x16x32_f16(false, (A), false, (B), (short)0, (C), false, false)

namespace cfg {
constexpr int BATCH = 8;
constexpr int N     = 1024;
constexpr int C     = 512;
constexpr int H     = 8;
constexpr int D     = 64;          // C / H
constexpr int BH    = BATCH * H;   // 64
constexpr int C3    = 3 * C;       // 1536
constexpr int ROWS  = BATCH * N;   // 8192
constexpr float EPS = 1e-6f;

// workspace offsets (in _Float16 elements); all 16B-aligned
constexpr size_t XH     = 0;                          // [ROWS][C]
constexpr size_t WQKVT  = XH     + (size_t)ROWS * C;  // [C3][C]   (w_qkv^T)
constexpr size_t WOUTT  = WQKVT  + (size_t)C3 * C;    // [C][C]    (w_out^T)
constexpr size_t MASKT  = WOUTT  + (size_t)C * C;     // [N][N]    (mask^T)
constexpr size_t QHT    = MASKT  + (size_t)N * N;     // [BH][D][N] each below
constexpr size_t KHT    = QHT + (size_t)BH * D * N;
constexpr size_t VHT    = KHT + (size_t)BH * D * N;
constexpr size_t QD     = VHT + (size_t)BH * D * N;   // [BH][N][D]
constexpr size_t KD     = QD  + (size_t)BH * N * D;   // [BH][N][D]
constexpr size_t OH     = KD  + (size_t)BH * N * D;   // [ROWS][C]
} // namespace cfg

// --------------------------- fragment loader -------------------------------
// 16-bit 16x32 fragment (ISA 7.12.2): lane L holds row r = L&15, K base
// kb = (L>>4)*8. VGPRs 0..3 hold K = kb..kb+7 (contiguous), VGPRs 4..7 hold
// K = kb+16..kb+23 (contiguous) => a fragment is two 16-byte loads.
// B mirrors A with lane = output column, so B operands are loaded with the
// same routine from a source laid out as [out_col][k] (i.e. B^T row-major).
__device__ __forceinline__ v16h load_frag(const _Float16* p, int ld) {
  const int lane = threadIdx.x & 31;
  const _Float16* row = p + (size_t)(lane & 15) * ld + ((lane >> 4) << 3);
  v8h lo, hi;
  __builtin_memcpy(&lo, row, 16);       // K = kb .. kb+7
  __builtin_memcpy(&hi, row + 16, 16);  // K = kb+16 .. kb+23
  return __builtin_shufflevector(lo, hi, 0, 1, 2, 3, 4, 5, 6, 7,
                                 8, 9, 10, 11, 12, 13, 14, 15);
}

// ------------------- async global -> LDS staging (gfx1250) -----------------
// 16 bytes per thread per call; completion via ASYNCcnt + workgroup barrier.
__device__ __forceinline__ void stage16(const _Float16* g, _Float16* l) {
#if __has_builtin(__builtin_amdgcn_global_load_async_to_lds_b128)
  __builtin_amdgcn_global_load_async_to_lds_b128(
      (gv4i*)(void*)g, (sv4i*)(void*)l, 0, 0);
#else
  v8h t;
  __builtin_memcpy(&t, g, 16);   // global_load_b128
  __builtin_memcpy(l, &t, 16);   // ds_store_b128
#endif
}

__device__ __forceinline__ void stage_wait() {
#if __has_builtin(__builtin_amdgcn_global_load_async_to_lds_b128)
#if __has_builtin(__builtin_amdgcn_s_wait_asynccnt)
  __builtin_amdgcn_s_wait_asynccnt(0);
#else
  asm volatile("s_wait_asynccnt 0" ::: "memory");
#endif
#endif
}

// ------------------------------ stage 0 ------------------------------------
__global__ void cvt_f32_to_f16_kernel(const float* __restrict__ src,
                                      _Float16* __restrict__ dst, int n) {
  int i = blockIdx.x * blockDim.x + threadIdx.x;
  if (i < n) dst[i] = (_Float16)src[i];
}

// dst[c][r] = (f16) src[r][c]; dst has shape [cols_src][rows_src]
__global__ void transpose_cvt_kernel(const float* __restrict__ src,
                                     _Float16* __restrict__ dst,
                                     int rows_src, int cols_src) {
  int i = blockIdx.x * blockDim.x + threadIdx.x;
  if (i < rows_src * cols_src) {
    int c = i / rows_src, r = i - c * rows_src;
    dst[i] = (_Float16)src[(size_t)r * cols_src + c];
  }
}

// ------------------------------ stage 1: QKV -------------------------------
// tiles: 512 (rows/16) x 96 (1536/16); one 16x16 tile per wave.
__global__ __launch_bounds__(256) void qkv_gemm_kernel(
    const _Float16* __restrict__ xh, const _Float16* __restrict__ wT,
    _Float16* __restrict__ ws) {
  const int wave = (blockIdx.x * blockDim.x + threadIdx.x) >> 5;
  const int mt = wave / (cfg::C3 / 16);
  const int nt = wave % (cfg::C3 / 16);
  const int lane = threadIdx.x & 31;

  const _Float16* ap = xh + (size_t)mt * 16 * cfg::C;
  const _Float16* bp = wT + (size_t)nt * 16 * cfg::C;

  v8f acc = {};
#pragma unroll 4
  for (int k0 = 0; k0 < cfg::C; k0 += 32) {
    if (k0 + 32 < cfg::C) {  // near prefetch of next K tile
      __builtin_prefetch(ap + (size_t)(lane & 15) * cfg::C + k0 + 32, 0, 3);
      __builtin_prefetch(bp + (size_t)(lane & 15) * cfg::C + k0 + 32, 0, 3);
    }
    acc = WMMA_F16(load_frag(ap + k0, cfg::C), load_frag(bp + k0, cfg::C), acc);
  }

  // epilogue: scatter into TRANSPOSED head layout [bh][d][n]; relu+eps on q/k
  const int c    = lane & 15;
  const int cg   = nt * 16 + c;
  const int part = cg >> 9;          // 0=q 1=k 2=v (uniform per tile)
  const int cl   = cg & (cfg::C - 1);
  const int h    = cl >> 6;
  const int d    = cl & (cfg::D - 1);
  _Float16* dst = ws + (part == 0 ? cfg::QHT : part == 1 ? cfg::KHT : cfg::VHT);
#pragma unroll
  for (int v = 0; v < 8; ++v) {
    const int r  = v + ((lane >> 4) << 3);
    const int rg = mt * 16 + r;            // global row in [0, 8192)
    const int b_ = rg >> 10;
    const int n  = rg & (cfg::N - 1);
    float val = acc[v];
    if (part < 2) val = fmaxf(val, 0.0f) + cfg::EPS;
    dst[((size_t)(b_ * cfg::H + h) << 16) + ((size_t)d << 10) + n] = (_Float16)val;
  }
}

// --------------------------- stage 2: diffusion ----------------------------
// Dst[bh][n][d] = Src[bh][n][d] + 0.1 * sum_m maskT[n][m] * Src[bh][m][d]
// A = maskT (row-major), B consumed from transposed Src^T [d][n] (row-major).
__global__ __launch_bounds__(256) void diffusion_kernel(_Float16* __restrict__ ws) {
  const int wave   = (blockIdx.x * blockDim.x + threadIdx.x) >> 5;
  const int tensor = wave >> 14;              // 16384 tiles per tensor
  int rem = wave & 16383;
  const int bh = rem >> 8;
  rem &= 255;
  const int mt = rem >> 2;                    // n tile
  const int nt = rem & 3;                     // d tile
  const int lane = threadIdx.x & 31;

  const _Float16* maskT = ws + cfg::MASKT;
  const _Float16* srcT = ws + (tensor ? cfg::KHT : cfg::QHT) + ((size_t)bh << 16);
  _Float16*       dst  = ws + (tensor ? cfg::KD  : cfg::QD)  + ((size_t)bh << 16);

  const _Float16* ap = maskT + (size_t)mt * 16 * cfg::N;
  const _Float16* bp = srcT + (size_t)nt * 16 * cfg::N;

  v8f acc = {};
#pragma unroll 2
  for (int k0 = 0; k0 < cfg::N; k0 += 32) {
    if (k0 + 32 < cfg::N) {
      __builtin_prefetch(ap + (size_t)(lane & 15) * cfg::N + k0 + 32, 0, 3);
      __builtin_prefetch(bp + (size_t)(lane & 15) * cfg::N + k0 + 32, 0, 3);
    }
    acc = WMMA_F16(load_frag(ap + k0, cfg::N), load_frag(bp + k0, cfg::N), acc);
  }

  const int c = lane & 15;
#pragma unroll
  for (int v = 0; v < 8; ++v) {
    const int r = v + ((lane >> 4) << 3);
    const int n = mt * 16 + r, d = nt * 16 + c;
    const float s = (float)srcT[(size_t)d * cfg::N + n];
    dst[(size_t)n * cfg::D + d] = (_Float16)(s + 0.1f * acc[v]);
  }
}

// ---------------------- stage 3: fused masked attention --------------------
// 8 waves/block, all on the SAME (b,h); per 32-m step, K[32][64] and
// V^T[64][32] tiles are staged into LDS once per block (async copies), then
// each wave computes its own 16-row n-tile flash-style.
__global__ __launch_bounds__(256) void attention_kernel(
    const float* __restrict__ mask, _Float16* __restrict__ ws) {
  __shared__ _Float16 ldsK[32 * 64];          // [m][d] tile, ld=64
  __shared__ _Float16 ldsV[64 * 32];          // [d][m] tile, ld=32
  __shared__ _Float16 smem_all[8][16 * 32];   // per-wave S staging tiles
  const int tid  = threadIdx.x;
  const int lane = tid & 31;
  const int wid  = tid >> 5;
  const int bh   = blockIdx.x >> 3;                  // 64 bh, 8 blocks each
  const int nt   = ((blockIdx.x & 7) << 3) + wid;    // 16-row n-tile

  const _Float16* Qp = ws + cfg::QD  + ((size_t)bh << 16);  // [n][d]
  const _Float16* Kp = ws + cfg::KD  + ((size_t)bh << 16);  // [m][d]
  const _Float16* VT = ws + cfg::VHT + ((size_t)bh << 16);  // [d][n]
  _Float16* smem = smem_all[wid];

  // Q row-tile fragments (rows n, K = d), loaded once per wave.
  const v16h qa0 = load_frag(Qp + (size_t)nt * 16 * cfg::D, cfg::D);
  const v16h qa1 = load_frag(Qp + (size_t)nt * 16 * cfg::D + 32, cfg::D);

  v8f acc0 = {}, acc1 = {}, acc2 = {}, acc3 = {};
  float psum[8];
#pragma unroll
  for (int v = 0; v < 8; ++v) psum[v] = 0.0f;

  const int c  = lane & 15;
  const int rh = (lane >> 4) << 3;

  for (int m0 = 0; m0 < cfg::N; m0 += 32) {
    // ---- cooperative staging of K tile (4KB) and V^T tile (4KB) ----
    // K: 32 rows x 128B; thread t copies halves [t*8, t*8+8).
    stage16(Kp + (size_t)m0 * cfg::D + tid * 8, ldsK + tid * 8);
    // V^T: 64 rows x 64B; thread t -> row t>>2, col halves (t&3)*8.
    stage16(VT + (size_t)(tid >> 2) * cfg::N + m0 + (tid & 3) * 8,
            ldsV + tid * 8);
    stage_wait();
    __syncthreads();

    // ---- S = Q . K^T for 32 columns (two 16-wide tiles); B from ldsK ----
    v8f s0 = {}, s1 = {};
    s0 = WMMA_F16(qa0, load_frag(ldsK, 64), s0);
    s0 = WMMA_F16(qa1, load_frag(ldsK + 32, 64), s0);
    s1 = WMMA_F16(qa0, load_frag(ldsK + 16 * 64, 64), s1);
    s1 = WMMA_F16(qa1, load_frag(ldsK + 16 * 64 + 32, 64), s1);

    // ---- elementwise mask, row-sum partials, transpose via LDS ----
#pragma unroll
    for (int v = 0; v < 8; ++v) {
      const int r = v + rh;
      const size_t mrow = (size_t)(nt * 16 + r) * cfg::N + m0;
      const float e0 = s0[v] * mask[mrow + c];
      const float e1 = s1[v] * mask[mrow + 16 + c];
      psum[v] += e0 + e1;
      smem[r * 32 + c]      = (_Float16)e0;
      smem[r * 32 + 16 + c] = (_Float16)e1;
    }
    __builtin_amdgcn_wave_barrier();  // DS ops are in-order within a wave

    // ---- acc += S_masked @ V[m0:m0+32]; B from ldsV ([d][m], ld=32) ----
    const v16h sa = load_frag(smem, 32);
    acc0 = WMMA_F16(sa, load_frag(ldsV +  0 * 32, 32), acc0);
    acc1 = WMMA_F16(sa, load_frag(ldsV + 16 * 32, 32), acc1);
    acc2 = WMMA_F16(sa, load_frag(ldsV + 32 * 32, 32), acc2);
    acc3 = WMMA_F16(sa, load_frag(ldsV + 48 * 32, 32), acc3);
    __syncthreads();  // protect ldsK/ldsV before next iteration's staging
  }

  // ---- reduce row sums across the 16 lanes holding each row's columns ----
#pragma unroll
  for (int v = 0; v < 8; ++v) {
#pragma unroll
    for (int m = 1; m < 16; m <<= 1) psum[v] += __shfl_xor(psum[v], m, 32);
    psum[v] = 1.0f / (psum[v] + cfg::EPS);   // z for row (v + rh)
  }

  // ---- write O in [B,N,C] f16 layout for the output projection ----
  _Float16* Oh = ws + cfg::OH;
  const int b_ = bh >> 3, h = bh & 7;
  const v8f* accs[4] = {&acc0, &acc1, &acc2, &acc3};
#pragma unroll
  for (int t = 0; t < 4; ++t) {
#pragma unroll
    for (int v = 0; v < 8; ++v) {
      const int r = v + rh;
      const size_t row = (size_t)(b_ * cfg::N + nt * 16 + r);
      Oh[row * cfg::C + h * cfg::D + t * 16 + c] = (_Float16)((*accs[t])[v] * psum[v]);
    }
  }
}

// ----------------------- stage 4: output projection ------------------------
// tiles: 512 (rows/16) x 32 (512/16); B from w_out^T row-major.
__global__ __launch_bounds__(256) void out_gemm_kernel(
    const _Float16* __restrict__ oh, const _Float16* __restrict__ wT,
    const float* __restrict__ b_out, float* __restrict__ out) {
  const int wave = (blockIdx.x * blockDim.x + threadIdx.x) >> 5;
  const int mt = wave >> 5;
  const int nt = wave & 31;
  const int lane = threadIdx.x & 31;

  const _Float16* ap = oh + (size_t)mt * 16 * cfg::C;
  const _Float16* bp = wT + (size_t)nt * 16 * cfg::C;

  v8f acc = {};
#pragma unroll 4
  for (int k0 = 0; k0 < cfg::C; k0 += 32) {
    if (k0 + 32 < cfg::C) {
      __builtin_prefetch(ap + (size_t)(lane & 15) * cfg::C + k0 + 32, 0, 3);
      __builtin_prefetch(bp + (size_t)(lane & 15) * cfg::C + k0 + 32, 0, 3);
    }
    acc = WMMA_F16(load_frag(ap + k0, cfg::C), load_frag(bp + k0, cfg::C), acc);
  }

  const int c    = lane & 15;
  const int cg   = nt * 16 + c;
  const float bias = b_out[cg];
#pragma unroll
  for (int v = 0; v < 8; ++v) {
    const int rg = mt * 16 + v + ((lane >> 4) << 3);
    out[(size_t)rg * cfg::C + cg] = acc[v] + bias;
  }
}

// ------------------------------- launcher ----------------------------------
extern "C" void kernel_launch(void* const* d_in, const int* in_sizes, int n_in,
                              void* d_out, int out_size, void* d_ws, size_t ws_size,
                              hipStream_t stream) {
  using namespace cfg;
  const float* x     = (const float*)d_in[0];  // [B,N,C]
  const float* mask  = (const float*)d_in[1];  // [N,N]
  const float* w_qkv = (const float*)d_in[2];  // [C,3C]
  const float* w_out = (const float*)d_in[3];  // [C,C]
  const float* b_out = (const float*)d_in[4];  // [C]
  float* out = (float*)d_out;                  // [B,N,C]
  _Float16* ws = (_Float16*)d_ws;

  // stage 0: f16 conversions + transposed copies
  {
    int n = ROWS * C;
    cvt_f32_to_f16_kernel<<<(n + 255) / 256, 256, 0, stream>>>(x, ws + XH, n);
    n = C * C3;  // w_qkv [C][C3] -> [C3][C]
    transpose_cvt_kernel<<<(n + 255) / 256, 256, 0, stream>>>(w_qkv, ws + WQKVT, C, C3);
    n = C * C;   // w_out [C][C] -> w_out^T
    transpose_cvt_kernel<<<(n + 255) / 256, 256, 0, stream>>>(w_out, ws + WOUTT, C, C);
    n = N * N;   // mask [m][n] -> maskT [n][m]
    transpose_cvt_kernel<<<(n + 255) / 256, 256, 0, stream>>>(mask, ws + MASKT, N, N);
  }

  // stage 1: qkv GEMM -> transposed head-layout Q/K/V (relu+eps on q,k)
  {
    const int tiles = (ROWS / 16) * (C3 / 16);   // 512*96
    qkv_gemm_kernel<<<tiles / 8, 256, 0, stream>>>(ws + XH, ws + WQKVT, ws);
  }

  // stage 2: graph diffusion for Q and K  (-> row-major Qd/Kd)
  {
    const int tiles = 2 * BH * (N / 16) * (D / 16); // 32768
    diffusion_kernel<<<tiles / 8, 256, 0, stream>>>(ws);
  }

  // stage 3: fused masked linear attention (8 waves share one (b,h) per block)
  {
    attention_kernel<<<BH * 8, 256, 0, stream>>>(mask, ws);
  }

  // stage 4: output projection + bias
  {
    const int tiles = (ROWS / 16) * (C / 16); // 16384
    out_gemm_kernel<<<tiles / 8, 256, 0, stream>>>(ws + OH, ws + WOUTT, b_out, out);
  }
}
